// LineProposalNetwork_62182536511764
// MI455X (gfx1250) — matrix operand order repaired and checked by reading, not compiled, not run on hardware
//
#include <hip/hip_runtime.h>
#include <cstdint>

typedef __attribute__((ext_vector_type(16))) _Float16 v16h;
typedef __attribute__((ext_vector_type(8)))  _Float16 v8h;
typedef __attribute__((ext_vector_type(8)))  float    v8f;

#define HW_ (512*512)

union AFrag { v16h v; v8h h[2]; _Float16 e[16]; };

__device__ inline unsigned monof(float f) {
  unsigned u = __float_as_uint(f);
  return (u & 0x80000000u) ? ~u : (u | 0x80000000u);
}
__device__ inline float unmonof(unsigned m) {
  return (m & 0x80000000u) ? __uint_as_float(m ^ 0x80000000u) : __uint_as_float(~m);
}
__device__ inline float sigmoidf_(float x) { return 1.0f / (1.0f + expf(-x)); }

// One 16B global->LDS async copy; ldsOff = low 32 bits of flat shared address.
__device__ inline void asyncCopy16(unsigned ldsOff, const void* gptr) {
  asm volatile("global_load_async_to_lds_b128 %0, %1, off"
               :: "v"(ldsOff), "v"(gptr) : "memory");
}
__device__ inline void asyncWait0() {
  asm volatile("s_wait_asynccnt 0x0" ::: "memory");
}

// Map global map-channel (0..10) + batch n to flat d_out offset (floats).
__device__ inline size_t mapOffset(int gch, int n) {
  const size_t HWc = (size_t)HW_;
  if (gch == 0) return (size_t)n * HWc;                       // lmap
  if (gch == 1) return 2*HWc + (size_t)n * HWc;               // jmap
  if (gch <= 3) return 4*HWc + (size_t)(n*2 + (gch-2)) * HWc; // joff
  if (gch == 4) return 8*HWc + (size_t)n * HWc;               // cmap
  if (gch <= 6) return 10*HWc + (size_t)(n*2 + (gch-5)) * HWc;// coff
  return 14*HWc + (size_t)(n*4 + (gch-7)) * HWc;              // eoff
}

// ---------------- features NCHW f32 -> NHWC f16 ----------------
__global__ __launch_bounds__(256) void prep_nhwc(const float* __restrict__ feat,
                                                 _Float16* __restrict__ out) {
  long long e = (long long)blockIdx.x * 256 + threadIdx.x;
  if (e >= 2ll*64*HW_) return;
  int ch = (int)(e & 63);
  long long p = e >> 6;
  int x = (int)(p & 511);
  int y = (int)((p >> 9) & 511);
  int n = (int)(p >> 18);
  out[e] = (_Float16)feat[(((size_t)(n*64 + ch))*512 + y)*512 + x];
}

__global__ void zero_stats(float* stats) { if (threadIdx.x < 64) stats[threadIdx.x] = 0.f; }

// ---------------- conv1: implicit GEMM, WMMA f16->f32 ----------------
__global__ __launch_bounds__(256) void conv1_wmma(const _Float16* __restrict__ feat,
                                                  const float* __restrict__ w1,
                                                  const float* __restrict__ b1,
                                                  int head, float* __restrict__ hOut) {
  __shared__ _Float16 Bs[18*32*16];   // 18KB: B fragments per K-chunk in lane layout
  const int tid = threadIdx.x;
  for (int t = tid; t < 18*32*16; t += 256) {
    int kc = t >> 9, rem = t & 511;
    int lane = rem >> 4, i = rem & 15;
    int n = lane & 15;
    int k = kc*32 + ((i < 8) ? (lane >> 4)*8 + i : 16 + (lane >> 4)*8 + (i - 8));   // B layout: table order (8h+s | 16+8h+s-8)
    int tap = k >> 6, c = k & 63;
    Bs[t] = (_Float16)w1[((((size_t)(head*16 + n))*64 + c)*3 + tap/3)*3 + (tap%3)];
  }
  __syncthreads();

  const int wv   = tid >> 5;
  const int lane = tid & 31;
  const int m    = lane & 15;
  const int base = (lane >> 4) * 8;
  const int y = blockIdx.y, n_img = blockIdx.z;
  const int x = blockIdx.x*128 + wv*16 + m;

  float bias = b1[head*16 + (lane & 15)];
  v8f acc;
#pragma unroll
  for (int v = 0; v < 8; ++v) acc[v] = bias;

  for (int kc = 0; kc < 18; ++kc) {
    int tap = kc >> 1;
    int c0  = (kc & 1) * 32;
    int yy = y + tap/3 - 1, xx = x + tap%3 - 1;
    AFrag a;
    if ((unsigned)yy < 512u && (unsigned)xx < 512u) {
      const _Float16* p = feat + ((size_t)((n_img*512 + yy)*512 + xx))*64 + c0 + base;
      __builtin_prefetch(p + 64, 0, 1);
      a.h[0] = *(const v8h*)p;
      a.h[1] = *(const v8h*)(p + 16);
    } else {
#pragma unroll
      for (int i = 0; i < 16; ++i) a.e[i] = (_Float16)0.f;
    }
    union { v16h v; v8h h[2]; } bf;
    const _Float16* bp = &Bs[(kc*32 + lane)*16];
    bf.h[0] = *(const v8h*)bp;
    bf.h[1] = *(const v8h*)(bp + 8);
    acc = __builtin_amdgcn_wmma_f32_16x16x32_f16(false, a.v, false, bf.v,
                                                 (short)0, acc, false, false);
  }
  const int px0 = blockIdx.x*128 + wv*16;
  const int oc  = lane & 15;
#pragma unroll
  for (int v = 0; v < 8; ++v) {
    int xo = px0 + v + base;   // D: M = v + half*8
    hOut[((size_t)((n_img*512 + y)*512 + xo))*16 + oc] = acc[v];
  }
}

// ---------------- BN statistics over h (NHWC, C=16) ----------------
__global__ __launch_bounds__(256) void bn_stats(const float* __restrict__ h,
                                                float* __restrict__ stats) {
  __shared__ float ssum[16], ssq[16];
  int tid = threadIdx.x;
  if (tid < 16) { ssum[tid] = 0.f; ssq[tid] = 0.f; }
  __syncthreads();
  float s = 0.f, q = 0.f;
  size_t stride = (size_t)gridDim.x * blockDim.x;      // multiple of 16
  for (size_t e = (size_t)blockIdx.x*blockDim.x + tid; e < 8388608ull; e += stride) {
    float v = h[e]; s += v; q += v*v;
  }
  atomicAdd(&ssum[tid & 15], s);
  atomicAdd(&ssq [tid & 15], q);
  __syncthreads();
  if (tid < 16) { atomicAdd(&stats[tid], ssum[tid]); atomicAdd(&stats[16+tid], ssq[tid]); }
}

__global__ void bn_finalize(float* stats, const float* __restrict__ gamma,
                            const float* __restrict__ beta, int head) {
  int c = threadIdx.x;
  if (c >= 16) return;
  const float inv_n = 1.0f / 524288.0f;
  float mean = stats[c] * inv_n;
  float var  = stats[16+c] * inv_n - mean*mean;
  float inv  = rsqrtf(var + 1e-5f);
  float sc   = gamma[head*16 + c] * inv;
  stats[32+c] = sc;
  stats[48+c] = beta[head*16 + c] - mean*sc;
}

// ---------------- conv2: BN+ReLU fused into A-fragment, WMMA ----------------
__global__ __launch_bounds__(256) void conv2_wmma(const float* __restrict__ hIn,
                                                  const float* __restrict__ w2,
                                                  const float* __restrict__ b2,
                                                  const float* __restrict__ stats,
                                                  int head, int nch, int gch0, int dosig,
                                                  float* __restrict__ out) {
  __shared__ _Float16 Bs[5*32*16];
  __shared__ __align__(16) float bnSB[32];   // [0..15]=scale, [16..31]=shift
  int tid = threadIdx.x;
  // Async DMA the 32 BN coefficients (stats[32..63]) into LDS: 8 x 16B.
  if (tid < 8) {
    unsigned ldsOff = (unsigned)(size_t)(&bnSB[tid*4]);
    asyncCopy16(ldsOff, (const void*)(stats + 32 + tid*4));
  }
  for (int t = tid; t < 5*32*16; t += 256) {
    int kc = t >> 9, rem = t & 511;
    int lane = rem >> 4, i = rem & 15;
    int n = lane & 15;
    int k = kc*32 + ((i < 8) ? (lane >> 4)*8 + i : 16 + (lane >> 4)*8 + (i - 8));
    int tap = k >> 4, c = k & 15;
    float wv = 0.f;
    if (tap < 9 && n < 4)
      wv = w2[((((size_t)(head*4 + n))*16 + c)*3 + tap/3)*3 + (tap%3)];
    Bs[t] = (_Float16)wv;
  }
  asyncWait0();
  __syncthreads();

  const int wvid = tid >> 5;
  const int lane = tid & 31;
  const int m    = lane & 15;
  const int base = (lane >> 4) * 8;
  const int y = blockIdx.y, n_img = blockIdx.z;
  const int x = blockIdx.x*128 + wvid*16 + m;
  const int oc = lane & 15;

  float bias = (oc < 4) ? b2[head*4 + oc] : 0.f;
  v8f acc;
#pragma unroll
  for (int v = 0; v < 8; ++v) acc[v] = bias;

  for (int kc = 0; kc < 5; ++kc) {
    AFrag a;
#pragma unroll
    for (int half8 = 0; half8 < 2; ++half8) {
      int tp = kc*2 + half8;
      int yy = y + tp/3 - 1, xx = x + tp%3 - 1;
      bool ok = (tp < 9) && ((unsigned)yy < 512u) && ((unsigned)xx < 512u);
      if (ok) {
        const float* p = hIn + ((size_t)((n_img*512 + yy)*512 + xx))*16 + base;
#pragma unroll
        for (int i = 0; i < 8; ++i) {
          float hv = fmaxf(p[i]*bnSB[base+i] + bnSB[16+base+i], 0.f);
          a.e[half8*8 + i] = (_Float16)hv;
        }
      } else {
#pragma unroll
        for (int i = 0; i < 8; ++i) a.e[half8*8 + i] = (_Float16)0.f;
      }
    }
    union { v16h v; v8h h[2]; } bf;
    const _Float16* bp = &Bs[(kc*32 + lane)*16];
    bf.h[0] = *(const v8h*)bp;
    bf.h[1] = *(const v8h*)(bp + 8);
    acc = __builtin_amdgcn_wmma_f32_16x16x32_f16(false, a.v, false, bf.v,
                                                 (short)0, acc, false, false);
  }
  if (oc < nch) {
    size_t ob = mapOffset(gch0 + oc, n_img);
    const int px0 = blockIdx.x*128 + wvid*16;
#pragma unroll
    for (int v = 0; v < 8; ++v) {
      int xo = px0 + v + base;
      float val = acc[v];
      if (dosig) val = sigmoidf_(val);
      out[ob + (size_t)y*512 + xo] = val;
    }
  }
}

// ---------------- key builders: (desc score | asc index) packed u64 ----------------
__global__ __launch_bounds__(256) void build_keys_j(const float* __restrict__ jm,
                                                    unsigned long long* __restrict__ keys) {
  int idx = blockIdx.x * 256 + threadIdx.x;
  if (idx >= HW_) return;
  int y = idx >> 9, x = idx & 511;
  float v = jm[idx];
  float mx = -3.4e38f;
  for (int dy = -1; dy <= 1; ++dy) {
    int yy = y + dy; if ((unsigned)yy >= 512u) continue;
    for (int dx = -1; dx <= 1; ++dx) {
      int xx = x + dx; if ((unsigned)xx >= 512u) continue;
      mx = fmaxf(mx, jm[yy*512 + xx]);
    }
  }
  float jv = (v == mx) ? v : 0.0f;
  keys[idx] = (((unsigned long long)(~monof(jv))) << 32) | (unsigned)idx;
}

__global__ __launch_bounds__(256) void build_keys_c(const float* __restrict__ cm,
                                                    unsigned long long* __restrict__ keys) {
  int idx = blockIdx.x * 256 + threadIdx.x;
  if (idx >= HW_) return;
  keys[idx] = (((unsigned long long)(~monof(cm[idx]))) << 32) | (unsigned)idx;
}

// ---------------- global bitonic sort, 2^18 u64, one workgroup ----------------
__global__ __launch_bounds__(1024) void bitonic_global(unsigned long long* __restrict__ d) {
  const int n = HW_;
  for (int k = 2; k <= n; k <<= 1)
    for (int j = k >> 1; j > 0; j >>= 1) {
      for (int i = threadIdx.x; i < n; i += 1024) {
        int ixj = i ^ j;
        if (ixj > i) {
          unsigned long long a = d[i], b = d[ixj];
          bool up = ((i & k) == 0);
          if ((a > b) == up) { d[i] = b; d[ixj] = a; }
        }
      }
      __syncthreads();
    }
}

__device__ void sortShared8192(unsigned long long* sk, int tid) {
  for (int k = 2; k <= 8192; k <<= 1)
    for (int j = k >> 1; j > 0; j >>= 1) {
      for (int i = tid; i < 8192; i += 1024) {
        int ixj = i ^ j;
        if (ixj > i) {
          unsigned long long a = sk[i], b = sk[ixj];
          bool up = ((i & k) == 0);
          if ((a > b) == up) { sk[i] = b; sk[ixj] = a; }
        }
      }
      __syncthreads();
    }
}

// ---------------- decode ----------------
__global__ __launch_bounds__(256) void decode_junc(const unsigned long long* __restrict__ keys,
                                                   const float* __restrict__ joff0,
                                                   const float* __restrict__ joff1,
                                                   float* __restrict__ junc) {
  int j = blockIdx.x * 256 + threadIdx.x;
  if (j >= 300) return;
  unsigned idx = (unsigned)keys[j];
  float jx = (float)(idx & 511), jy = (float)(idx >> 9);
  const float CL = 511.9999f;
  junc[2*j+0] = fminf(fmaxf(jx + joff0[idx] + 0.5f, 0.f), CL);
  junc[2*j+1] = fminf(fmaxf(jy + joff1[idx] + 0.5f, 0.f), CL);
}

__global__ __launch_bounds__(256) void decode_lines(
    const unsigned long long* __restrict__ keys,
    const float* __restrict__ coff0, const float* __restrict__ coff1,
    const float* __restrict__ e0c, const float* __restrict__ e1c,
    const float* __restrict__ e2c, const float* __restrict__ e3c,
    const float* __restrict__ junc,
    float* __restrict__ loi, float* __restrict__ lsc, float* __restrict__ cost,
    unsigned* __restrict__ imin, unsigned* __restrict__ imax, unsigned* __restrict__ keep) {
  __shared__ __align__(16) float sj[600];
  // Async DMA the 600-float junction table into LDS: 150 x 16B.
  for (int t = threadIdx.x; t < 150; t += 256) {
    unsigned ldsOff = (unsigned)(size_t)(&sj[t*4]);
    asyncCopy16(ldsOff, (const void*)(junc + t*4));
  }
  asyncWait0();
  __syncthreads();
  int r = blockIdx.x * 256 + threadIdx.x;
  if (r >= 5000) return;
  unsigned long long key = keys[r];
  unsigned idx = (unsigned)key;
  float score = unmonof(~(unsigned)(key >> 32));
  float cx = (float)(idx & 511) + coff0[idx] + 0.5f;
  float cy = (float)(idx >> 9)  + coff1[idx] + 0.5f;
  const float CL = 511.9999f;
  float p0x = fminf(fmaxf(cx + e0c[idx], 0.f), CL);
  float p0y = fminf(fmaxf(cy + e1c[idx], 0.f), CL);
  float p1x = fminf(fmaxf(cx, 0.f), CL);
  float p1y = fminf(fmaxf(cy, 0.f), CL);
  float p2x = fminf(fmaxf(cx + e2c[idx], 0.f), CL);
  float p2y = fminf(fmaxf(cy + e3c[idx], 0.f), CL);
  loi[r*6+0]=p0x; loi[r*6+1]=p0y; loi[r*6+2]=p1x;
  loi[r*6+3]=p1y; loi[r*6+4]=p2x; loi[r*6+5]=p2y;
  lsc[r] = score;
  float b1v = 3.4e38f, b2v = 3.4e38f;
  unsigned i1 = 0, i2 = 0;
  for (unsigned j = 0; j < 300; ++j) {
    float dx = p0x - sj[2*j], dy = p0y - sj[2*j+1];
    float d = dx*dx + dy*dy;
    if (d < b1v) { b1v = d; i1 = j; }
    dx = p2x - sj[2*j]; dy = p2y - sj[2*j+1];
    d = dx*dx + dy*dy;
    if (d < b2v) { b2v = d; i2 = j; }
  }
  unsigned mi = (i1 < i2) ? i1 : i2;
  unsigned ma = (i1 < i2) ? i2 : i1;
  imin[r] = mi; imax[r] = ma; keep[r] = (mi != ma) ? 1u : 0u;
  cost[r] = b1v + b2v - logf(score);
}

__global__ __launch_bounds__(1024) void decode_sortsel(
    const float* __restrict__ cost, const unsigned* __restrict__ imin,
    const unsigned* __restrict__ imax, const unsigned* __restrict__ keep,
    unsigned* __restrict__ ordc, unsigned* __restrict__ keyv,
    unsigned* __restrict__ fm, unsigned* __restrict__ validp,
    unsigned* __restrict__ rfinal, unsigned* __restrict__ vout) {
  __shared__ unsigned long long sk[8192];   // 64KB
  int tid = threadIdx.x;
  // sort by cost (stable via index in low bits)
  for (int p = tid; p < 8192; p += 1024)
    sk[p] = (p < 5000) ? ((((unsigned long long)monof(cost[p])) << 32) | (unsigned)p)
                       : 0xFFFFFFFFFFFFFFFFull;
  __syncthreads();
  sortShared8192(sk, tid);
  for (int p = tid; p < 5000; p += 1024) ordc[p] = (unsigned)sk[p];
  __syncthreads();
  // sort by junction-pair key (dedup prep)
  for (int p = tid; p < 8192; p += 1024) {
    unsigned long long kk = 0xFFFFFFFFFFFFFFFFull;
    if (p < 5000) {
      unsigned r = ordc[p];
      unsigned kv = keep[r] ? (imin[r]*300u + imax[r]) : 90000u;
      keyv[p] = kv;
      kk = (((unsigned long long)kv) << 32) | (unsigned)p;
    }
    sk[p] = kk;
  }
  __syncthreads();
  sortShared8192(sk, tid);
  for (int s = tid; s < 5000; s += 1024) {
    unsigned p = (unsigned)sk[s];
    unsigned kcur = (unsigned)(sk[s] >> 32);
    fm[p] = (s == 0) || ((unsigned)(sk[s-1] >> 32) != kcur);
  }
  __syncthreads();
  // final sort: uniques first in junction-pair-key order, rest in cost order
  for (int p = tid; p < 8192; p += 1024) {
    unsigned long long kk = 0xFFFFFFFFFFFFFFFFull;
    if (p < 5000) {
      unsigned r = ordc[p];
      unsigned vv = (fm[p] && keep[r]) ? 1u : 0u;
      validp[p] = vv;
      unsigned k2 = vv ? keyv[p] : 90001u;
      kk = (((unsigned long long)k2) << 32) | (unsigned)p;
    }
    sk[p] = kk;
  }
  __syncthreads();
  sortShared8192(sk, tid);
  for (int q = tid; q < 5000; q += 1024) {
    unsigned p = (unsigned)sk[q];
    rfinal[q] = ordc[p];
    vout[q] = validp[p];
  }
}

__global__ __launch_bounds__(256) void decode_write(
    const unsigned* __restrict__ rfin, const unsigned* __restrict__ vout,
    const float* __restrict__ loi, const float* __restrict__ lsc,
    const unsigned* __restrict__ imin, const unsigned* __restrict__ imax,
    const float* __restrict__ junc, float* __restrict__ out, int n) {
  int q = blockIdx.x * 256 + threadIdx.x;
  if (q >= 5000) return;
  unsigned r = rfin[q];
  float p0x = loi[r*6+0], p0y = loi[r*6+1];
  float p1x = loi[r*6+2], p1y = loi[r*6+3];
  float p2x = loi[r*6+4], p2y = loi[r*6+5];
  unsigned mi = imin[r], ma = imax[r];
  float e0x = junc[2*mi], e0y = junc[2*mi+1];
  float e1x = junc[2*ma], e1y = junc[2*ma+1];
  if (e0y > e1y) { float t = e0x; e0x = e1x; e1x = t; t = e0y; e0y = e1y; e1y = t; }
  float d0x = e0x - p0x, d0y = e0y - p0y;
  float d1x = e1x - p2x, d1y = e1y - p2y;
  size_t b = 5767168ull + ((size_t)(n*5000 + q))*6ull;
  out[b+0] = p0x + d0x;              out[b+1] = p0y + d0y;
  out[b+2] = p1x + 0.5f*(d0x+d1x);   out[b+3] = p1y + 0.5f*(d0y+d1y);
  out[b+4] = p2x + d1x;              out[b+5] = p2y + d1y;
  out[5827168ull + (size_t)(n*5000 + q)] = lsc[r];
  out[5837168ull + (size_t)(n*5000 + q)] = vout[q] ? 1.0f : 0.0f;
}

// ---------------- host ----------------
extern "C" void kernel_launch(void* const* d_in, const int* in_sizes, int n_in,
                              void* d_out, int out_size, void* d_ws, size_t ws_size,
                              hipStream_t stream) {
  (void)in_sizes; (void)n_in; (void)out_size; (void)ws_size;
  const float* features = (const float*)d_in[0];
  const float* w1    = (const float*)d_in[1];
  const float* b1    = (const float*)d_in[2];
  const float* gamma = (const float*)d_in[3];
  const float* beta  = (const float*)d_in[4];
  const float* w2    = (const float*)d_in[5];
  const float* b2    = (const float*)d_in[6];
  float* out = (float*)d_out;

  char* ws = (char*)d_ws;
  _Float16* featN = (_Float16*)ws;                                 // 64 MB
  float* hBuf  = (float*)(ws + 67108864ull);                       // 32 MB
  float* stats = (float*)(ws + 100663296ull);                      // 1 KB pad
  unsigned long long* keys = (unsigned long long*)(ws + 100664320ull); // 2 MB
  char* dp = ws + 102761472ull;
  float* junc = (float*)dp;     dp += 4800;
  float* loi  = (float*)dp;     dp += 120000;
  float* lsc  = (float*)dp;     dp += 20000;
  float* cost = (float*)dp;     dp += 20000;
  unsigned* uimin = (unsigned*)dp; dp += 20000;
  unsigned* uimax = (unsigned*)dp; dp += 20000;
  unsigned* ukeep = (unsigned*)dp; dp += 20000;
  unsigned* ordc  = (unsigned*)dp; dp += 20000;
  unsigned* ukeyv = (unsigned*)dp; dp += 20000;
  unsigned* ufm   = (unsigned*)dp; dp += 20000;
  unsigned* uvalp = (unsigned*)dp; dp += 20000;
  unsigned* urfin = (unsigned*)dp; dp += 20000;
  unsigned* uvout = (unsigned*)dp; dp += 20000;

  prep_nhwc<<<131072, 256, 0, stream>>>(features, featN);

  static const int gch0s[6] = {0,1,2,4,5,7};
  static const int nchs[6]  = {1,1,2,1,2,4};
  static const int sigs[6]  = {1,1,0,1,0,0};
  for (int head = 0; head < 6; ++head) {
    zero_stats<<<1, 64, 0, stream>>>(stats);
    conv1_wmma<<<dim3(4,512,2), 256, 0, stream>>>(featN, w1, b1, head, hBuf);
    bn_stats<<<1024, 256, 0, stream>>>(hBuf, stats);
    bn_finalize<<<1, 16, 0, stream>>>(stats, gamma, beta, head);
    conv2_wmma<<<dim3(4,512,2), 256, 0, stream>>>(hBuf, w2, b2, stats, head,
                                                  nchs[head], gch0s[head], sigs[head], out);
  }

  const size_t HWs = (size_t)HW_;
  for (int n = 0; n < 2; ++n) {
    const float* jmapP = out + 2*HWs + (size_t)n*HWs;
    const float* joff0 = out + 4*HWs + (size_t)(n*2+0)*HWs;
    const float* joff1 = out + 4*HWs + (size_t)(n*2+1)*HWs;
    const float* cmapP = out + 8*HWs + (size_t)n*HWs;
    const float* coff0 = out + 10*HWs + (size_t)(n*2+0)*HWs;
    const float* coff1 = out + 10*HWs + (size_t)(n*2+1)*HWs;
    const float* e0c = out + 14*HWs + (size_t)(n*4+0)*HWs;
    const float* e1c = out + 14*HWs + (size_t)(n*4+1)*HWs;
    const float* e2c = out + 14*HWs + (size_t)(n*4+2)*HWs;
    const float* e3c = out + 14*HWs + (size_t)(n*4+3)*HWs;

    build_keys_j<<<1024, 256, 0, stream>>>(jmapP, keys);
    bitonic_global<<<1, 1024, 0, stream>>>(keys);
    decode_junc<<<2, 256, 0, stream>>>(keys, joff0, joff1, junc);
    build_keys_c<<<1024, 256, 0, stream>>>(cmapP, keys);
    bitonic_global<<<1, 1024, 0, stream>>>(keys);
    decode_lines<<<20, 256, 0, stream>>>(keys, coff0, coff1, e0c, e1c, e2c, e3c,
                                         junc, loi, lsc, cost, uimin, uimax, ukeep);
    decode_sortsel<<<1, 1024, 0, stream>>>(cost, uimin, uimax, ukeep,
                                           ordc, ukeyv, ufm, uvalp, urfin, uvout);
    decode_write<<<20, 256, 0, stream>>>(urfin, uvout, loi, lsc, uimin, uimax, junc, out, n);
  }
}
